// AxialAttention_72035191489149
// MI455X (gfx1250) — matrix operand order repaired and checked by reading, not compiled, not run on hardware
//
#include <hip/hip_runtime.h>

// ---------------------------------------------------------------------------
// AxialAttention for MI455X (gfx1250, wave32, WMMA bf16 -> f32)
// Pipeline:
//   P1: pack w_qkv (f32) -> WMMA A-fragment layout (bf16)
//   P2: pack/transpose x -> WMMA B-fragment layout (bf16), via LDS tiles
//   G : qkv GEMM (v_wmma_f32_16x16x32_bf16) + per-(n,o) BN partial stats
//   R1: reduce qkv BN stats -> per-channel scale/shift
//   S : per-(n,g) recompute qk (WMMA) / qr / kr, per-WG sim BN partials
//   R2: reduce sim BN stats -> per-channel (3x8) scale/shift
//   A : attention: qk(WMMA)+qr+kr -> BN-combined -> softmax -> sv(WMMA)+sve
//       + per-channel out BN partials
//   R3: reduce out BN stats
//   F : out BN + pair-sum + transpose to (B, OUT, T, H, W)
// ---------------------------------------------------------------------------

typedef __attribute__((ext_vector_type(16))) __bf16 v16bf;
typedef __attribute__((ext_vector_type(8)))  float  v8f;

#define NB   4
#define CIN  512
#define NT   4
#define HH   56
#define WW   56
#define NG   8
#define NN   896          /* B*T*W */
#define NOUT 1024         /* 2*G*GP */
#define RELW 111          /* 2K-1 */
#define EPSF 1e-5f

// ---------------- workspace layout (bytes) ----------------
static constexpr size_t algn(size_t x) { return (x + 255) & ~size_t(255); }
static constexpr size_t SZ_WF  = (size_t)64 * 16 * 32 * 16 * 2;   // 1 MB   packed W
static constexpr size_t SZ_XF  = (size_t)NN * 16 * 4 * 512 * 2;   // 56 MB  packed X
static constexpr size_t SZ_QKV = (size_t)NN * NOUT * HH * 2;      // 98 MB  qkv raw (bf16)
static constexpr size_t SZ_QP  = (size_t)2 * NN * NOUT * 4;       // 7 MB   qkv BN partials
static constexpr size_t SZ_QAB = (size_t)2 * NOUT * 4;
static constexpr size_t SZ_SP  = (size_t)2 * 3 * NN * NG * 4;     // sim BN partials
static constexpr size_t SZ_SAB = (size_t)48 * 4;
static constexpr size_t SZ_OR  = SZ_QKV;                          // out raw (bf16)
static constexpr size_t SZ_OP  = SZ_QP;
static constexpr size_t SZ_OAB = SZ_QAB;

static constexpr size_t OFF_WF  = 0;
static constexpr size_t OFF_XF  = algn(OFF_WF  + SZ_WF);
static constexpr size_t OFF_QKV = algn(OFF_XF  + SZ_XF);
static constexpr size_t OFF_QP  = algn(OFF_QKV + SZ_QKV);
static constexpr size_t OFF_QAB = algn(OFF_QP  + SZ_QP);
static constexpr size_t OFF_SP  = algn(OFF_QAB + SZ_QAB);
static constexpr size_t OFF_SAB = algn(OFF_SP  + SZ_SP);
static constexpr size_t OFF_OR  = algn(OFF_SAB + SZ_SAB);
static constexpr size_t OFF_OP  = algn(OFF_OR  + SZ_OR);
static constexpr size_t OFF_OAB = algn(OFF_OP  + SZ_OP);

// ---------------- small helpers ----------------
__device__ inline float wred32(float v) {
  #pragma unroll
  for (int m = 1; m < 32; m <<= 1) v += __shfl_xor(v, m);
  return v;
}
__device__ inline float hred16(float v) {
  #pragma unroll
  for (int m = 1; m < 16; m <<= 1) v += __shfl_xor(v, m);
  return v;
}

__device__ inline v8f wmma_bf16(v16bf a, v16bf b, v8f c) {
  return __builtin_amdgcn_wmma_f32_16x16x32_bf16(false, a, false, b, (short)0, c,
                                                 false, false);
}

// A-fragment (16x32 bf16), source is transposed: A[m][k] = M[k*ld + i0 + m]
__device__ inline v16bf frag_a_T32(const float* M, int ld, int i0, int lane) {
  int m = lane & 15, half = lane >> 4;
  v16bf a;
  #pragma unroll
  for (int e = 0; e < 8; ++e) a[e]     = (__bf16)M[(half * 16 + e) * ld + i0 + m];
  #pragma unroll
  for (int e = 0; e < 8; ++e) a[e + 8] = (__bf16)M[(half * 16 + 8 + e) * ld + i0 + m];
  return a;
}
// A-fragment, row-major source: A[m][k] = M[(r0+m)*ld + k0 + k]
__device__ inline v16bf frag_a_row(const float* M, int ld, int r0, int k0, int lane) {
  int m = lane & 15, half = lane >> 4;
  const float* row = M + (r0 + m) * ld + k0;
  v16bf a;
  #pragma unroll
  for (int e = 0; e < 8; ++e) a[e]     = (__bf16)row[half * 16 + e];
  #pragma unroll
  for (int e = 0; e < 8; ++e) a[e + 8] = (__bf16)row[half * 16 + 8 + e];
  return a;
}
// B-fragment (32x16 bf16), K-major source: B[k][n] = M[k*ld + j0 + n]
__device__ inline v16bf frag_b_colmaj(const float* M, int ld, int j0, int lane) {
  int col = lane & 15, kh = lane >> 4;
  v16bf b;
  #pragma unroll
  for (int e = 0; e < 16; ++e) b[e] = (__bf16)M[(kh * 16 + e) * ld + j0 + col];
  return b;
}
// B-fragment, N-major source: B[k][n] = M[(j0+n)*ld + k0 + k]
__device__ inline v16bf frag_b_rowmaj(const float* M, int ld, int j0, int k0, int lane) {
  int col = lane & 15, kh = lane >> 4;
  const float* row = M + (j0 + col) * ld + k0 + kh * 16;
  v16bf b;
  #pragma unroll
  for (int e = 0; e < 16; ++e) b[e] = (__bf16)row[e];
  return b;
}

// ---------------- P1: pack W into A-fragment layout ----------------
__global__ void k_pack_w(const float* __restrict__ w, __bf16* __restrict__ wf) {
  int idx = blockIdx.x * 256 + threadIdx.x;          // over 1024*512
  if (idx >= 1024 * 512) return;
  int o = idx >> 9, c = idx & 511;
  int ot = o >> 4, m = o & 15;
  int kb = c >> 5, kin = c & 31;
  int half, e;
  half = kin >> 4; e = kin & 15;
  int lane = half * 16 + m;
  wf[((size_t)(ot * 16 + kb) * 32 + lane) * 16 + e] = (__bf16)w[idx];
}

// ---------------- P2: pack/transpose x into B-fragment layout ----------------
__global__ void k_pack_x(const float* __restrict__ x, __bf16* __restrict__ xf) {
  // grid.x = cblk*4 + wt (64), grid.y = t, grid.z = b
  int cblk = blockIdx.x >> 2, wt = blockIdx.x & 3;
  int t = blockIdx.y, b = blockIdx.z;
  const int NWT = 14;
  __shared__ __bf16 lx[32 * 56 * NWT];               // [ci][h][wi]
  for (int idx = threadIdx.x; idx < 32 * 56 * NWT; idx += 256) {
    int ci = idx / (56 * NWT);
    int rem = idx % (56 * NWT);
    int h = rem / NWT, wi = rem % NWT;
    float v = x[(((size_t)(b * 512 + cblk * 32 + ci)) * 4 + t) * 3136 +
                h * 56 + (wt * NWT + wi)];
    lx[idx] = (__bf16)v;
  }
  __syncthreads();
  for (int idx = threadIdx.x; idx < NWT * 2048; idx += 256) {
    int wi = idx >> 11;
    int q  = idx & 2047;
    int nt = q >> 9, r = q & 511;
    int lane = r >> 4, e = r & 15;
    int col = lane & 15, kh = lane >> 4;
    int h   = nt * 16 + col;
    int kin = kh * 16 + e;
    __bf16 v = (h < 56) ? lx[kin * (56 * NWT) + h * NWT + wi] : (__bf16)0.f;
    int n = b * 224 + t * 56 + wt * NWT + wi;
    xf[((size_t)(n * 16 + cblk) * 4 + nt) * 512 + r] = v;
  }
}

// ---------------- G: qkv GEMM (WMMA) + BN partials ----------------
__global__ void k_qkv_gemm(const __bf16* __restrict__ wf, const __bf16* __restrict__ xf,
                           __bf16* __restrict__ qkv, float* __restrict__ qpart) {
  int n = blockIdx.x, ob = blockIdx.y;               // ob 0..7 -> 128 rows
  int wv = threadIdx.x >> 5, lane = threadIdx.x & 31;
  int ot = ob * 8 + wv;                              // 0..63
  v8f acc[4];
  v8f vz = {0.f, 0.f, 0.f, 0.f, 0.f, 0.f, 0.f, 0.f};
  #pragma unroll
  for (int nt = 0; nt < 4; ++nt) acc[nt] = vz;

  const v16bf* A = (const v16bf*)wf + (size_t)(ot * 16) * 32 + lane;
  const v16bf* Bx = (const v16bf*)xf + (size_t)(n * 16) * 4 * 32 + lane;
  #pragma unroll 4
  for (int kb = 0; kb < 16; ++kb) {
    __builtin_prefetch(A + (kb + 1) * 32, 0, 1);
    __builtin_prefetch(Bx + (kb + 1) * 128, 0, 1);
    v16bf a = A[(size_t)kb * 32];
    #pragma unroll
    for (int nt = 0; nt < 4; ++nt) {
      v16bf b = Bx[(size_t)(kb * 4 + nt) * 32];
      acc[nt] = wmma_bf16(a, b, acc[nt]);
    }
  }

  int col = lane & 15, half = lane >> 4;
  int obase = ob * 128 + wv * 16;
  float s8[8], q8[8];
  #pragma unroll
  for (int r = 0; r < 8; ++r) { s8[r] = 0.f; q8[r] = 0.f; }
  #pragma unroll
  for (int nt = 0; nt < 4; ++nt) {
    int h = nt * 16 + col;
    if (h < 56) {
      #pragma unroll
      for (int r = 0; r < 8; ++r) {
        float v = acc[nt][r];
        int orow = obase + r + half * 8;
        qkv[((size_t)n * NOUT + orow) * HH + h] = (__bf16)v;
        s8[r] += v; q8[r] += v * v;
      }
    }
  }
  #pragma unroll
  for (int r = 0; r < 8; ++r) {
    float ss = hred16(s8[r]);
    float qq = hred16(q8[r]);
    if (col == 0) {
      int orow = obase + r + half * 8;
      qpart[(size_t)n * NOUT + orow] = ss;
      qpart[(size_t)NN * NOUT + (size_t)n * NOUT + orow] = qq;
    }
  }
}

// ---------------- R1/R3: per-channel BN coefficient reduce ----------------
__global__ void k_bn_scale(const float* __restrict__ part, const float* __restrict__ gamma,
                           const float* __restrict__ beta, float* __restrict__ ab) {
  int o = blockIdx.x * blockDim.x + threadIdx.x;
  if (o >= NOUT) return;
  float s = 0.f, q = 0.f;
  for (int n = 0; n < NN; ++n) {
    s += part[(size_t)n * NOUT + o];
    q += part[(size_t)NN * NOUT + (size_t)n * NOUT + o];
  }
  const float inv = 1.f / (NN * 56.f);
  float mean = s * inv;
  float var  = q * inv - mean * mean;
  float a = gamma[o] * rsqrtf(var + EPSF);
  ab[o] = a;
  ab[NOUT + o] = beta[o] - mean * a;
}

// load BN-applied q,k tiles into LDS (padded to 64 cols with zeros)
__device__ inline void load_qk_lds(const __bf16* qkv, const float* ab, int n, int g,
                                   float* qn, float* kn) {
  for (int idx = threadIdx.x; idx < 2 * 32 * 64; idx += 256) {
    int which = idx >> 11;
    int r = idx & 2047;
    int c = r >> 6, h = r & 63;
    int o = g * 128 + which * 32 + c;
    float v = 0.f;
    if (h < 56) v = (float)qkv[((size_t)n * NOUT + o) * HH + h] * ab[o] + ab[NOUT + o];
    (which ? kn : qn)[c * 64 + h] = v;
  }
}

// relative-position dot over 32 channels
__device__ inline float dot_rel(const float* mat, const float* __restrict__ rel,
                                int rowbase, int i, int ridx) {
  float s = 0.f;
  #pragma unroll 8
  for (int c = 0; c < 32; ++c)
    s = fmaf(mat[c * 64 + i], rel[(rowbase + c) * RELW + ridx], s);
  return s;
}

__device__ inline v8f wmma_qk_tile(const float* qn, const float* kn, int i0, int j0,
                                   int lane) {
  v16bf a = frag_a_T32(qn, 64, i0, lane);
  v16bf b = frag_b_colmaj(kn, 64, j0, lane);
  v8f z = {0.f, 0.f, 0.f, 0.f, 0.f, 0.f, 0.f, 0.f};
  return wmma_bf16(a, b, z);
}

// ---------------- S: sim BN statistics ----------------
__global__ void k_sim_stats(const __bf16* __restrict__ qkv, const float* __restrict__ ab,
                            const float* __restrict__ rel, float* __restrict__ spart) {
  int n = blockIdx.x, g = blockIdx.y;
  __shared__ float qn[32 * 64], kn[32 * 64];
  __shared__ float red[8][6];
  load_qk_lds(qkv, ab, n, g, qn, kn);
  __syncthreads();
  int wv = threadIdx.x >> 5, lane = threadIdx.x & 31;
  int col = lane & 15, half = lane >> 4;

  // qk stats via WMMA (each wave: 2 of the 16 tiles)
  float qs = 0.f, qq = 0.f;
  int mt = wv >> 1;
  #pragma unroll
  for (int kk = 0; kk < 2; ++kk) {
    int ntl = (wv & 1) * 2 + kk;
    v8f acc = wmma_qk_tile(qn, kn, mt * 16, ntl * 16, lane);
    int j = ntl * 16 + col;
    if (j < 56) {
      #pragma unroll
      for (int r = 0; r < 8; ++r) {
        int i = mt * 16 + r + half * 8;
        if (i < 56) { float v = acc[r]; qs += v; qq += v * v; }
      }
    }
  }
  // qr / kr stats via VALU
  float rs = 0.f, rq = 0.f, ks = 0.f, kq = 0.f;
  for (int idx = threadIdx.x; idx < 56 * 56; idx += 256) {
    int i = idx / 56, j = idx % 56;
    float vq = dot_rel(qn, rel, 0, i, i - j + 55);
    rs += vq; rq += vq * vq;
    float vk = dot_rel(kn, rel, 32, i, i - j + 55);   // stats invariant to transpose
    ks += vk; kq += vk * vk;
  }
  float vals[6] = {qs, qq, rs, rq, ks, kq};
  #pragma unroll
  for (int t = 0; t < 6; ++t) vals[t] = wred32(vals[t]);
  if (lane == 0) {
    #pragma unroll
    for (int t = 0; t < 6; ++t) red[wv][t] = vals[t];
  }
  __syncthreads();
  if (threadIdx.x == 0) {
    float tot[6] = {0.f, 0.f, 0.f, 0.f, 0.f, 0.f};
    for (int w = 0; w < 8; ++w)
      for (int t = 0; t < 6; ++t) tot[t] += red[w][t];
    int ng = n * NG + g;
    const int NC = NN * NG;
    spart[0 * NC + ng] = tot[0];
    spart[1 * NC + ng] = tot[2];
    spart[2 * NC + ng] = tot[4];
    spart[3 * NC + 0 * NC + ng] = tot[1];
    spart[3 * NC + 1 * NC + ng] = tot[3];
    spart[3 * NC + 2 * NC + ng] = tot[5];
  }
}

// ---------------- R2: sim BN coefficients (24 channels) ----------------
__global__ void k_bn_sim(const float* __restrict__ spart, const float* __restrict__ gs,
                         const float* __restrict__ bs, float* __restrict__ sab) {
  int ch = threadIdx.x;
  if (ch >= 24) return;
  int comp = ch >> 3, g = ch & 7;
  const int NC = NN * NG;
  float s = 0.f, q = 0.f;
  for (int n = 0; n < NN; ++n) {
    int ng = n * NG + g;
    s += spart[comp * NC + ng];
    q += spart[3 * NC + comp * NC + ng];
  }
  const float inv = 1.f / (NN * 3136.f);
  float mean = s * inv;
  float var  = q * inv - mean * mean;
  float a = gs[ch] * rsqrtf(var + EPSF);
  sab[ch] = a;
  sab[24 + ch] = bs[ch] - mean * a;
}

// ---------------- A: attention ----------------
__global__ void k_attn(const __bf16* __restrict__ qkv, const float* __restrict__ ab,
                       const float* __restrict__ rel, const float* __restrict__ sab,
                       __bf16* __restrict__ oraw, float* __restrict__ opart) {
  int n = blockIdx.x, g = blockIdx.y;
  __shared__ float qn[32 * 64], kn[32 * 64], V[64 * 64], S[64 * 64];
  int tid = threadIdx.x, wv = tid >> 5, lane = tid & 31;
  int col = lane & 15, half = lane >> 4;

  load_qk_lds(qkv, ab, n, g, qn, kn);
  for (int idx = tid; idx < 64 * 64; idx += 256) {
    int c = idx >> 6, h = idx & 63;
    int o = g * 128 + 64 + c;
    float v = 0.f;
    if (h < 56) v = (float)qkv[((size_t)n * NOUT + o) * HH + h] * ab[o] + ab[NOUT + o];
    V[idx] = v;
  }
  float a0 = sab[g], a1 = sab[8 + g], a2 = sab[16 + g];
  float btot = sab[24 + g] + sab[24 + 8 + g] + sab[24 + 16 + g];
  __syncthreads();

  // base: a1*qr + a2*kr^T + btot (VALU), zero padding elsewhere
  for (int idx = tid; idx < 64 * 64; idx += 256) {
    int i = idx >> 6, j = idx & 63;
    float v = 0.f;
    if (i < 56 && j < 56)
      v = a1 * dot_rel(qn, rel, 0, i, i - j + 55) +
          a2 * dot_rel(kn, rel, 32, j, j - i + 55) + btot;
    S[idx] = v;
  }
  __syncthreads();

  // + a0*qk via WMMA (each wave owns 2 disjoint tiles)
  {
    int mt = wv >> 1;
    #pragma unroll
    for (int kk = 0; kk < 2; ++kk) {
      int ntl = (wv & 1) * 2 + kk;
      v8f acc = wmma_qk_tile(qn, kn, mt * 16, ntl * 16, lane);
      #pragma unroll
      for (int r = 0; r < 8; ++r)
        S[(mt * 16 + r + half * 8) * 64 + ntl * 16 + col] += a0 * acc[r];
    }
  }
  __syncthreads();

  // softmax over j, in place; cols 56..63 forced to zero
  for (int i = wv; i < 56; i += 8) {
    float v0 = S[i * 64 + lane];
    float v1 = (lane < 24) ? S[i * 64 + 32 + lane] : -3.0e38f;
    float mx = fmaxf(v0, v1);
    #pragma unroll
    for (int m = 1; m < 32; m <<= 1) mx = fmaxf(mx, __shfl_xor(mx, m));
    float e0 = __expf(v0 - mx);
    float e1 = (lane < 24) ? __expf(v1 - mx) : 0.f;
    float sm = wred32(e0 + e1);
    float inv = 1.f / sm;
    S[i * 64 + lane] = e0 * inv;
    S[i * 64 + 32 + lane] = e1 * inv;
  }
  __syncthreads();

  if (wv < 4) {
    // sv = V @ P^T via WMMA: wave wv owns rows c in [wv*16, wv*16+16)
    int c0 = wv * 16;
    v8f acc[4];
    v8f vz = {0.f, 0.f, 0.f, 0.f, 0.f, 0.f, 0.f, 0.f};
    #pragma unroll
    for (int nt = 0; nt < 4; ++nt) acc[nt] = vz;
    #pragma unroll
    for (int ks = 0; ks < 2; ++ks) {
      v16bf afr = frag_a_row(V, 64, c0, ks * 32, lane);
      #pragma unroll
      for (int nt = 0; nt < 4; ++nt) {
        v16bf bfr = frag_b_rowmaj(S, 64, nt * 16, ks * 32, lane);
        acc[nt] = wmma_bf16(afr, bfr, acc[nt]);
      }
    }
    float s8[8], q8[8];
    #pragma unroll
    for (int r = 0; r < 8; ++r) { s8[r] = 0.f; q8[r] = 0.f; }
    #pragma unroll
    for (int nt = 0; nt < 4; ++nt) {
      int i = nt * 16 + col;
      if (i < 56) {
        #pragma unroll
        for (int r = 0; r < 8; ++r) {
          float v = acc[nt][r];
          int c = c0 + r + half * 8;
          oraw[((size_t)n * NOUT + g * 128 + 2 * c) * HH + i] = (__bf16)v;
          s8[r] += v; q8[r] += v * v;
        }
      }
    }
    #pragma unroll
    for (int r = 0; r < 8; ++r) {
      float ss = hred16(s8[r]);
      float qq = hred16(q8[r]);
      if (col == 0) {
        int c = c0 + r + half * 8;
        opart[(size_t)n * NOUT + g * 128 + 2 * c] = ss;
        opart[(size_t)NN * NOUT + (size_t)n * NOUT + g * 128 + 2 * c] = qq;
      }
    }
  } else {
    // sve[c,i] = sum_j P[i][j] * rel[64+c][i-j+55]  (waves 4..7)
    int w4 = wv - 4;
    for (int c2 = 0; c2 < 16; ++c2) {
      int c = w4 * 16 + c2;
      const float* rr = rel + (64 + c) * RELW;
      int i = lane;
      float acc0 = 0.f;
      #pragma unroll 8
      for (int j = 0; j < 56; ++j) acc0 = fmaf(S[i * 64 + j], rr[i - j + 55], acc0);
      int i2 = lane + 32;
      float acc1 = 0.f;
      if (i2 < 56) {
        #pragma unroll 8
        for (int j = 0; j < 56; ++j) acc1 = fmaf(S[i2 * 64 + j], rr[i2 - j + 55], acc1);
      }
      size_t base = ((size_t)n * NOUT + g * 128 + 2 * c + 1) * HH;
      oraw[base + i] = (__bf16)acc0;
      if (i2 < 56) oraw[base + i2] = (__bf16)acc1;
      float s = acc0 + ((i2 < 56) ? acc1 : 0.f);
      float q = acc0 * acc0 + ((i2 < 56) ? acc1 * acc1 : 0.f);
      s = wred32(s); q = wred32(q);
      if (lane == 0) {
        opart[(size_t)n * NOUT + g * 128 + 2 * c + 1] = s;
        opart[(size_t)NN * NOUT + (size_t)n * NOUT + g * 128 + 2 * c + 1] = q;
      }
    }
  }
}

// ---------------- F: final BN + pair-sum + transpose ----------------
__global__ void k_final(const __bf16* __restrict__ oraw, const float* __restrict__ oab,
                        float* __restrict__ out) {
  int og = blockIdx.x;          // 0..127 (groups of 4 output channels)
  int t = blockIdx.y, b = blockIdx.z;
  __shared__ float cmb[56 * 57];
  for (int oi = 0; oi < 4; ++oi) {
    int o = og * 4 + oi;
    float a0 = oab[2 * o],     c0 = oab[NOUT + 2 * o];
    float a1 = oab[2 * o + 1], c1 = oab[NOUT + 2 * o + 1];
    for (int idx = threadIdx.x; idx < 3136; idx += 256) {
      int w = idx / 56, h = idx % 56;
      int n = b * 224 + t * 56 + w;
      size_t base = ((size_t)n * NOUT + 2 * o) * HH + h;
      float r0 = (float)oraw[base];
      float r1 = (float)oraw[base + HH];
      cmb[w * 57 + h] = a0 * r0 + c0 + a1 * r1 + c1;
    }
    __syncthreads();
    for (int idx = threadIdx.x; idx < 3136; idx += 256) {
      int h = idx / 56, w = idx % 56;
      out[(((size_t)(b * 512 + o) * 4 + t) * 56 + h) * 56 + w] = cmb[w * 57 + h];
    }
    __syncthreads();
  }
}

// ---------------- launch ----------------
extern "C" void kernel_launch(void* const* d_in, const int* in_sizes, int n_in,
                              void* d_out, int out_size, void* d_ws, size_t ws_size,
                              hipStream_t stream) {
  const float* x      = (const float*)d_in[0];
  const float* w_qkv  = (const float*)d_in[1];
  const float* g_qkv  = (const float*)d_in[2];
  const float* b_qkv  = (const float*)d_in[3];
  const float* rel    = (const float*)d_in[4];
  const float* g_sim  = (const float*)d_in[5];
  const float* b_sim  = (const float*)d_in[6];
  const float* g_out  = (const float*)d_in[7];
  const float* b_out  = (const float*)d_in[8];

  char* ws = (char*)d_ws;
  __bf16* wf   = (__bf16*)(ws + OFF_WF);
  __bf16* xf   = (__bf16*)(ws + OFF_XF);
  __bf16* qkvb = (__bf16*)(ws + OFF_QKV);
  float*  qpart = (float*)(ws + OFF_QP);
  float*  qab   = (float*)(ws + OFF_QAB);
  float*  spart = (float*)(ws + OFF_SP);
  float*  sab   = (float*)(ws + OFF_SAB);
  __bf16* oraw = (__bf16*)(ws + OFF_OR);
  float*  opart = (float*)(ws + OFF_OP);
  float*  oab   = (float*)(ws + OFF_OAB);

  k_pack_w<<<2048, 256, 0, stream>>>(w_qkv, wf);
  k_pack_x<<<dim3(64, 4, 4), 256, 0, stream>>>(x, xf);
  k_qkv_gemm<<<dim3(NN, 8), 256, 0, stream>>>(wf, xf, qkvb, qpart);
  k_bn_scale<<<4, 256, 0, stream>>>(qpart, g_qkv, b_qkv, qab);
  k_sim_stats<<<dim3(NN, NG), 256, 0, stream>>>(qkvb, qab, rel, spart);
  k_bn_sim<<<1, 32, 0, stream>>>(spart, g_sim, b_sim, sab);
  k_attn<<<dim3(NN, NG), 256, 0, stream>>>(qkvb, qab, rel, sab, oraw, opart);
  k_bn_scale<<<4, 256, 0, stream>>>(opart, g_out, b_out, oab);
  k_final<<<dim3(128, 4, 4), 256, 0, stream>>>(oraw, oab, (float*)d_out);
}
